// LSTM_Network_65051574665514
// MI455X (gfx1250) — matrix operand order repaired and checked by reading, not compiled, not run on hardware
//
#include <hip/hip_runtime.h>

// ---------------------------------------------------------------------------
// 2-layer LSTM (B=256, T=1024, D=128, H=256) + dense head, for MI455X gfx1250.
// v3: - step kernel templated on KK so both GEMM phases have compile-time K:
//       fully unrolled K loops, shift-only index math, more WMMA pipelining.
//     - async DMA issue loop is now a uniform counted loop (no EXEC masking).
// Pipeline per step: GLOBAL_LOAD_ASYNC_TO_LDS_B128 (ASYNCcnt) DMAs the 32xK
// f32 A panel to LDS, one s_wait_asynccnt + barrier per phase, then a
// barrier-free K loop of v_wmma_f32_16x16x32_bf16 with bf16 weights (N-major,
// L2-resident) and in-register f32->bf16 A conversion.
// ---------------------------------------------------------------------------

typedef __bf16 v16bf __attribute__((ext_vector_type(16)));
typedef float  v8f   __attribute__((ext_vector_type(8)));

#define BDIM 256
#define HDIM 256
#define TDIM 1024
#define DDIM 128
#define APAD 268   // f32 panel row stride: 268%64=12 -> conflict-spread banks

__device__ inline __bf16 f2bf(float f) {
  unsigned u = __builtin_bit_cast(unsigned, f);
  unsigned r = u + 0x7FFFu + ((u >> 16) & 1u);  // round-to-nearest-even
  unsigned short h = (unsigned short)(r >> 16);
  return __builtin_bit_cast(__bf16, h);
}

__device__ inline float sigm(float x) { return 1.0f / (1.0f + __expf(-x)); }

// --- one-time: transpose + downconvert weights: src f32 [K][1024] -> dst bf16 [1024][K]
__global__ __launch_bounds__(256)
void wconv_kernel(const float* __restrict__ src, __bf16* __restrict__ dst, int K) {
  int e = blockIdx.x * 256 + threadIdx.x;   // e in [0, 1024*K)
  int n = e / K;
  int k = e - n * K;
  dst[(size_t)n * K + k] = f2bf(src[(size_t)k * 1024 + n]);
}

__global__ __launch_bounds__(256)
void zero_kernel(float* __restrict__ p, int n) {
  int i = blockIdx.x * 256 + threadIdx.x;
  if (i < n) p[i] = 0.0f;
}

// ---------------------------------------------------------------------------
// One LSTM layer step:  z = xin@Wk + hprev@Wr   (bias added in epilogue)
// Grid (8,4) x 256 threads; block = 32 rows x 64 cols of h; 8 waves:
// wave -> gate (w>>1), row-half (w&1); 4 x 16x16 accum tiles per wave.
// KK = K of the Wk phase (128 for layer 0, 256 for layer 1).
// ---------------------------------------------------------------------------
template <int KK>
__global__ __launch_bounds__(256)
void lstm_step_kernel(const float* __restrict__ xin, int ldx,
                      const __bf16* __restrict__ WkT, const __bf16* __restrict__ WrT,
                      const float* __restrict__ bias,
                      const float* __restrict__ hprev, const float* __restrict__ cprev,
                      float* __restrict__ hnew, float* __restrict__ cnew) {
  // f32 A panel and z-exchange buffer have disjoint lifetimes -> union them.
  __shared__ union SmemU {
    float a[32][APAD];      // async-DMA'd f32 A panel (<= 32 x 256 used)
    float z[4][32][64];     // gate exchange buffer (32 KB)
  } sm;

  const int tid   = threadIdx.x;
  const int lane  = tid & 31;
  const int wave  = tid >> 5;
  const int half  = lane >> 4;   // 0/1: lane group
  const int ln    = lane & 15;
  const int gate  = wave >> 1;   // 0..3 : i,f,g,o
  const int mhalf = wave & 1;    // which 16-row half of the 32-row block
  const int rowbase = blockIdx.x * 32;
  const int colbase = blockIdx.y * 64;

  const v8f vzero = {0.f, 0.f, 0.f, 0.f, 0.f, 0.f, 0.f, 0.f};
  v8f acc[4];
#pragma unroll
  for (int nt = 0; nt < 4; ++nt) acc[nt] = vzero;

#pragma unroll
  for (int ph = 0; ph < 2; ++ph) {
    const float*  Ap  = ph ? hprev : xin;
    const int     lda = ph ? HDIM  : ldx;
    const int     Kd  = ph ? HDIM  : KK;     // compile-time constant each phase
    const __bf16* WT  = ph ? WrT   : WkT;

    // --- async DMA the whole 32 x Kd f32 panel into LDS (ASYNCcnt path) ---
    __syncthreads();  // previous consumers of sm.a are done
    const int ncol4 = Kd >> 2;               // 16B chunks per row (const)
    const int nq    = (32 * ncol4) >> 8;     // chunks per thread (const)
#pragma unroll
    for (int q = 0; q < nq; ++q) {           // uniform counted loop: no EXEC mask
      const int c    = tid + (q << 8);
      const int row  = c / ncol4;            // const divisor -> shifts
      const int col4 = c - row * ncol4;
      const float* gsrc = Ap + (size_t)(rowbase + row) * lda + (col4 << 2);
      unsigned lofs = (unsigned)(uintptr_t)(&sm.a[row][col4 << 2]);
      asm volatile("global_load_async_to_lds_b128 %0, %1, off"
                   :: "v"(lofs), "v"(gsrc) : "memory");
    }
    asm volatile("s_wait_asynccnt 0x0" ::: "memory");
    __syncthreads();  // all waves' DMA complete -> panel visible

    // --- barrier-free, fully unrolled K loop over the panel ---
#pragma unroll
    for (int kb = 0; kb < Kd; kb += 32) {
      // A fragment (contiguous K order, same as B), f32->bf16 in-register:
      // pairs j<4 -> K = kb + half*16 + 2j ; j>=4 -> +8
      const float* Arow = &sm.a[mhalf * 16 + ln][kb];
      v16bf af;
#pragma unroll
      for (int j = 0; j < 4; ++j) {
        const int k0 = half * 16 + 2 * j;
        af[2 * j]     = f2bf(Arow[k0]);
        af[2 * j + 1] = f2bf(Arow[k0 + 1]);
        af[2 * j + 8] = f2bf(Arow[8 + k0]);
        af[2 * j + 9] = f2bf(Arow[8 + k0 + 1]);
      }

      // B fragments: lanes 0-15 K=kb..kb+15, lanes 16-31 K=kb+16..kb+31;
      // N = lane%16. WT is N-major -> one contiguous 32B load per lane.
#pragma unroll
      for (int nt = 0; nt < 4; ++nt) {
        const __bf16* bp = WT + (size_t)(gate * 256 + colbase + nt * 16 + ln) * Kd
                              + kb + half * 16;
        v16bf bfrag = *reinterpret_cast<const v16bf*>(bp);
        acc[nt] = __builtin_amdgcn_wmma_f32_16x16x32_bf16(
            /*neg_a=*/false, af, /*neg_b=*/false, bfrag,
            /*c_mod=*/(short)0, acc[nt], /*reuse_a=*/false, /*reuse_b=*/false);
      }

      if (kb + 32 < Kd) {  // folds at compile time
        __builtin_prefetch(WT + (size_t)(gate * 256 + colbase + ln) * Kd + kb + 32, 0, 3);
      }
    }
  }

  // spill z tiles to LDS (C/D layout: vgpr i, lane -> M = i + 8*half, N = ln)
  __syncthreads();  // all waves done reading sm.a before z overwrites the union
#pragma unroll
  for (int nt = 0; nt < 4; ++nt)
#pragma unroll
    for (int i = 0; i < 8; ++i)
      sm.z[gate][mhalf * 16 + half * 8 + i][nt * 16 + ln] = acc[nt][i];
  __syncthreads();

  // fused LSTM cell epilogue over the 32x64 block
#pragma unroll
  for (int e0 = 0; e0 < 2048; e0 += 256) {
    const int e = e0 + tid;
    const int r = e >> 6, c = e & 63;
    const int colH = colbase + c;
    const int row  = rowbase + r;
    const float iv = sm.z[0][r][c] + bias[           colH];
    const float fv = sm.z[1][r][c] + bias[HDIM     + colH];
    const float gv = sm.z[2][r][c] + bias[2 * HDIM + colH];
    const float ov = sm.z[3][r][c] + bias[3 * HDIM + colH];
    const float cp = cprev[(size_t)row * HDIM + colH];
    const float cn = sigm(fv) * cp + sigm(iv) * tanhf(gv);
    const float hn = sigm(ov) * tanhf(cn);
    cnew[(size_t)row * HDIM + colH] = cn;
    hnew[(size_t)row * HDIM + colH] = hn;
  }
}

// --- dense head on final hb: hid = relu(hb@dW0+db0); out = softmax(hid@dW1+db1)
__global__ __launch_bounds__(128)
void head_kernel(const float* __restrict__ hb,
                 const float* __restrict__ dW0, const float* __restrict__ db0,
                 const float* __restrict__ dW1, const float* __restrict__ db1,
                 float* __restrict__ out) {
  __shared__ float sh[HDIM];
  __shared__ float hid[128];
  __shared__ float logits[16];
  const int row = blockIdx.x;
  const int tid = threadIdx.x;

  for (int k = tid; k < HDIM; k += 128) sh[k] = hb[(size_t)row * HDIM + k];
  __syncthreads();

  float s = db0[tid];
  for (int k = 0; k < HDIM; ++k) s += sh[k] * dW0[(size_t)k * 128 + tid];
  hid[tid] = fmaxf(s, 0.0f);
  __syncthreads();

  if (tid < 16) {
    float l = db1[tid];
    for (int k = 0; k < 128; ++k) l += hid[k] * dW1[(size_t)k * 16 + tid];
    logits[tid] = l;
  }
  __syncthreads();

  if (tid == 0) {
    float mx = logits[0];
    for (int c = 1; c < 16; ++c) mx = fmaxf(mx, logits[c]);
    float sum = 0.0f;
    for (int c = 0; c < 16; ++c) { float e = __expf(logits[c] - mx); logits[c] = e; sum += e; }
    const float inv = 1.0f / sum;
    for (int c = 0; c < 16; ++c) out[(size_t)row * 16 + c] = logits[c] * inv;
  }
}

// ---------------------------------------------------------------------------
extern "C" void kernel_launch(void* const* d_in, const int* in_sizes, int n_in,
                              void* d_out, int out_size, void* d_ws, size_t ws_size,
                              hipStream_t stream) {
  (void)in_sizes; (void)n_in; (void)out_size; (void)ws_size;
  const float* x   = (const float*)d_in[0];
  const float* k0  = (const float*)d_in[1];
  const float* r0  = (const float*)d_in[2];
  const float* b0  = (const float*)d_in[3];
  const float* k1  = (const float*)d_in[4];
  const float* r1  = (const float*)d_in[5];
  const float* b1  = (const float*)d_in[6];
  const float* dW0 = (const float*)d_in[7];
  const float* db0 = (const float*)d_in[8];
  const float* dW1 = (const float*)d_in[9];
  const float* db1 = (const float*)d_in[10];

  // workspace layout
  __bf16* k0T = (__bf16*)d_ws;                 // [1024][128]
  __bf16* r0T = k0T + (size_t)1024 * 128;      // [1024][256]
  __bf16* k1T = r0T + (size_t)1024 * 256;
  __bf16* r1T = k1T + (size_t)1024 * 256;
  float*  st  = (float*)(r1T + (size_t)1024 * 256);
  const size_t S = (size_t)BDIM * HDIM;        // 65536 floats per state buffer
  const size_t G = 4 * S;                      // one double-buffer group
  float* hA[2] = { st + 0 * G + 0 * S, st + 1 * G + 0 * S };
  float* cA[2] = { st + 0 * G + 1 * S, st + 1 * G + 1 * S };
  float* hB[2] = { st + 0 * G + 2 * S, st + 1 * G + 2 * S };
  float* cB[2] = { st + 0 * G + 3 * S, st + 1 * G + 3 * S };

  // one-time prep: convert/transpose weights, zero initial state (group 0)
  wconv_kernel<<<(1024 * DDIM) / 256, 256, 0, stream>>>(k0, k0T, DDIM);
  wconv_kernel<<<(1024 * HDIM) / 256, 256, 0, stream>>>(r0, r0T, HDIM);
  wconv_kernel<<<(1024 * HDIM) / 256, 256, 0, stream>>>(k1, k1T, HDIM);
  wconv_kernel<<<(1024 * HDIM) / 256, 256, 0, stream>>>(r1, r1T, HDIM);
  zero_kernel<<<(int)(G + 255) / 256, 256, 0, stream>>>(st, (int)G);

  const dim3 grid(BDIM / 32, 4, 1);  // 8 x 4 blocks: 32-row x 64-col tiles of h
  int cur = 0;
  for (int t = 0; t < TDIM; ++t) {
    // layer 0: xin = x[:, t, :] (row stride T*D)
    lstm_step_kernel<DDIM><<<grid, 256, 0, stream>>>(
        x + (size_t)t * DDIM, TDIM * DDIM, k0T, r0T, b0,
        hA[cur], cA[cur], hA[1 - cur], cA[1 - cur]);
    // layer 1: xin = fresh hA (row stride H)
    lstm_step_kernel<HDIM><<<grid, 256, 0, stream>>>(
        hA[1 - cur], HDIM, k1T, r1T, b1,
        hB[cur], cB[cur], hB[1 - cur], cB[1 - cur]);
    cur ^= 1;
  }

  head_kernel<<<BDIM, 128, 0, stream>>>(hB[cur], dW0, db0, dW1, db1, (float*)d_out);
}
